// Eq_NLMP3_18013092840059
// MI455X (gfx1250) — matrix operand order repaired and checked by reading, not compiled, not run on hardware
//
#include <hip/hip_runtime.h>
#include <hip/hip_bf16.h>

typedef __attribute__((ext_vector_type(16))) _Float16 v16h;
typedef __attribute__((ext_vector_type(4)))  _Float16 v4h;
typedef __attribute__((ext_vector_type(8)))  float    v8f;

#define NN 50000
#define NE 800000

// ---------- WMMA fragment helpers (gfx1250, wave32) ----------
// All WMMA operands live in LDS pre-swizzled in fragment order:
// a tile (32 lanes x 16 f16) is 512 contiguous f16; lane's data at lane*16.
// A-matrix semantics: lane half=lane/16, M=lane%16; element j holds
//   k_local = 16*(j>>3) + 8*half + (j&7)
// B-matrix semantics: lane N=lane%16; element j holds k_local = 16*(j>>3) + 8*half + (j&7), as for A.

__device__ __forceinline__ v16h frag_a_swz(const _Float16* A, int ks) {
  return *(const v16h*)(A + (ks << 9) + (threadIdx.x & 31) * 16);
}
__device__ __forceinline__ v16h frag_b_swz(const _Float16* B, int KT, int ks, int nt) {
  return *(const v16h*)(B + ((nt * KT + ks) << 9) + (threadIdx.x & 31) * 16);
}

template<int KT>
__device__ __forceinline__ v8f mm_tile(const _Float16* A, const _Float16* B, int nt) {
  v8f acc = {};
#pragma unroll
  for (int ks = 0; ks < KT; ++ks)
    acc = __builtin_amdgcn_wmma_f32_16x16x32_f16(false, frag_a_swz(A, ks), false,
                                                 frag_b_swz(B, KT, ks, nt), (short)0,
                                                 acc, false, false);
  return acc;
}

// inverse map for writing element (row el, k) into A-swizzled storage
__device__ __forceinline__ int a_swz(int el, int k) {
  const int ks = k >> 5, kk = k & 31;
  const int half = (kk >> 3) & 1;
  const int j = ((kk >> 4) << 3) | (kk & 7);
  return (ks << 9) + ((half << 4) + el) * 16 + j;
}

// store 4 consecutive k's (c 4-aligned) -> contiguous in swizzled layout
__device__ __forceinline__ void store4h(_Float16* A, int el, int c,
                                        float x, float y, float z, float w) {
  *(v4h*)(A + a_swz(el, c)) = (v4h){(_Float16)x, (_Float16)y, (_Float16)z, (_Float16)w};
}

// cooperative fill of a B-swizzled weight (Ksrc x N source, zero-pad K to KT*32)
__device__ __forceinline__ void fill_b(_Float16* dst, const float* src, int Ksrc, int N,
                                       int KT, int NT, float scale) {
  const int total = KT * NT * 512;
  for (int i = threadIdx.x; i < total; i += blockDim.x) {
    const int tile = i >> 9, within = i & 511;
    const int lane = within >> 4, j = within & 15;
    const int ks = tile % KT, nt = tile / KT;
    const int k = ks * 32 + ((j >> 3) << 4) + ((lane >> 4) << 3) + (j & 7);
    const int n = nt * 16 + (lane & 15);
    dst[i] = (_Float16)((k < Ksrc) ? src[k * N + n] * scale : 0.f);
  }
}

__device__ __forceinline__ float reduce16(float x) {
  x += __shfl_xor(x, 1, 32);
  x += __shfl_xor(x, 2, 32);
  x += __shfl_xor(x, 4, 32);
  x += __shfl_xor(x, 8, 32);
  return x;
}

__device__ __forceinline__ float sgm(float x) {
  return __builtin_amdgcn_rcpf(1.0f + __expf(-x));
}

// ======================= EDGE PIPELINE KERNEL =======================
// 128 threads = 4 waves; each wave owns a 16-edge tile.
__global__ void __launch_bounds__(128) edge_kernel(
    const int* __restrict__ eidx,
    const float* __restrict__ hn_s, const float* __restrict__ hn_v,
    const float* __restrict__ he_s, const float* __restrict__ he_v,
    const float* __restrict__ fe_s, const float* __restrict__ fe_v,
    const float* __restrict__ fes,  const float* __restrict__ norm,
    const float* __restrict__ ev_Ws, const float* __restrict__ ev_Wv,
    const float* __restrict__ ev_Ls, const float* __restrict__ ev_Lv,
    const float* __restrict__ fc_W1, const float* __restrict__ fc_W2,
    const float* __restrict__ eu_Ws, const float* __restrict__ eu_Wv,
    const float* __restrict__ eu_Ls, const float* __restrict__ eu_Lv,
    float* __restrict__ o_he_s, float* __restrict__ o_he_v,
    float* __restrict__ nf_s,   float* __restrict__ nf_v) {
  extern __shared__ char smem[];
  // ---- block-shared f16 weights, B-swizzled (scales folded) ----
  _Float16* Wevs = (_Float16*)(smem);            // 96x192:  KT=3 NT=12 (36864 B)
  _Float16* Wevv = (_Float16*)(smem + 36864);    // 64x64:   KT=2 NT=4  (8192 B)
  _Float16* Weus = (_Float16*)(smem + 45056);    // 96x192
  _Float16* Weuv = (_Float16*)(smem + 81920);    // 64x64
  _Float16* Wfc1 = (_Float16*)(smem + 90112);    // 32x16:   KT=1 NT=1  (1024 B)
  _Float16* Wfc2 = (_Float16*)(smem + 91136);    // 32x96:   KT=1 NT=6  (6144 B)
  _Float16* Wels = (_Float16*)(smem + 97280);    // 128x32:  KT=4 NT=2  (8192 B)
  _Float16* Welv = (_Float16*)(smem + 105472);   // 64x16:   KT=2 NT=1  (2048 B)
  float*    Levs = (float*)(smem + 107520);      // 128
  float*    Levv = (float*)(smem + 108032);      // 64
  char* wavebase = smem + 108288;

  const int wid = threadIdx.x >> 5;
  const int lane = threadIdx.x & 31;
  const int n16 = lane & 15;
  const int half = lane >> 4;

  char* wa = wavebase + wid * 24320;
  _Float16* A_s  = (_Float16*)(wa);              // 3 tiles (16x96), persists
  _Float16* A_v  = (_Float16*)(wa + 3072);       // 3 comps x 2 tiles (16x64 padded)
  float*    gate = (float*)(wa + 9216);          // 16x64
  float*    wbuf = (float*)(wa + 13312);         // 16x96 f32 (fc output)
  _Float16* g2v  = (_Float16*)(wa + 13312);      // alias: 3 comps x 2 tiles
  _Float16* g2s  = (_Float16*)(wa + 19456);      // 4 tiles (16x128)
  _Float16* fesA = (_Float16*)(wa + 19456);      // alias (early): 1 tile
  _Float16* relA = (_Float16*)(wa + 19456 + 2048); // alias (early): 1 tile
  float* s1   = (float*)(wa + 23552);            // 16
  float* v1   = (float*)(wa + 23616);            // 16x3
  float* s2v  = (float*)(wa + 23808);            // 16
  float* v2v  = (float*)(wa + 23872);            // 16x3
  float* nrm  = (float*)(wa + 24064);            // 16
  int*   esrc = (int*)(wa + 24128);              // 16
  int*   edst = (int*)(wa + 24192);              // 16

  // ---- convert weights to f16 LDS tiles (scales folded) ----
  const float s96 = rsqrtf(96.f), s48 = rsqrtf(48.f);
  const float s128 = rsqrtf(128.f), s64 = rsqrtf(64.f);
  fill_b(Wevs, ev_Ws, 96, 192, 3, 12, s96);
  fill_b(Weus, eu_Ws, 96, 192, 3, 12, s96);
  fill_b(Wevv, ev_Wv, 48, 64, 2, 4, s48);
  fill_b(Weuv, eu_Wv, 48, 64, 2, 4, s48);
  fill_b(Wfc1, fc_W1, 16, 16, 1, 1, 0.25f);
  fill_b(Wfc2, fc_W2, 16, 96, 1, 6, 0.25f);
  fill_b(Wels, eu_Ls, 128, 32, 4, 2, s128);
  fill_b(Welv, eu_Lv, 64, 16, 2, 1, s64);
  for (int i = threadIdx.x; i < 128; i += blockDim.x) Levs[i] = ev_Ls[i] * s128;
  for (int i = threadIdx.x; i < 64; i += blockDim.x)  Levv[i] = ev_Lv[i] * s64;
  __syncthreads();

  const int e0 = blockIdx.x * 64 + wid * 16;
  if (lane < 16) {
    int e = e0 + lane; if (e >= NE) e = NE - 1;
    esrc[lane] = eidx[e];
    edst[lane] = eidx[NE + e];
    nrm[lane]  = norm[e];
    s2v[lane]  = fe_s[e];
    v2v[lane * 3 + 0] = fe_v[e * 3 + 0];
    v2v[lane * 3 + 1] = fe_v[e * 3 + 1];
    v2v[lane * 3 + 2] = fe_v[e * 3 + 2];
  }
  __syncthreads();

  // ---- stage s_cat = [he_s, hn_s[src], hn_s[dst]] (vectorized, swizzled) ----
  for (int idx = lane; idx < 128; idx += 32) {
    int el = idx >> 3, ch = idx & 7;
    int e = e0 + el; if (e >= NE) e = NE - 1;
    float4 v = ((const float4*)(he_s + e * 32))[ch];
    store4h(A_s, el, ch * 4, v.x, v.y, v.z, v.w);
  }
  for (int idx = lane; idx < 128; idx += 32) {
    int el = idx >> 3, ch = idx & 7;
    float4 v = ((const float4*)(hn_s + esrc[el] * 32))[ch];
    store4h(A_s, el, 32 + ch * 4, v.x, v.y, v.z, v.w);
  }
  for (int idx = lane; idx < 128; idx += 32) {
    int el = idx >> 3, ch = idx & 7;
    float4 v = ((const float4*)(hn_s + edst[el] * 32))[ch];
    store4h(A_s, el, 64 + ch * 4, v.x, v.y, v.z, v.w);
  }
  // ---- stage v_cat = [he_v, hn_v[src], hn_v[dst]] per (el, m): b96 reads ----
  for (int idx = lane; idx < 256; idx += 32) {
    int el = idx >> 4, m = idx & 15;
    int e = e0 + el; if (e >= NE) e = NE - 1;
    const float* p = he_v + e * 48 + m * 3;
    int sw = a_swz(el, m);
    A_v[sw] = (_Float16)p[0]; A_v[1024 + sw] = (_Float16)p[1]; A_v[2048 + sw] = (_Float16)p[2];
  }
  for (int idx = lane; idx < 256; idx += 32) {
    int el = idx >> 4, m = idx & 15;
    const float* p = hn_v + esrc[el] * 48 + m * 3;
    int sw = a_swz(el, 16 + m);
    A_v[sw] = (_Float16)p[0]; A_v[1024 + sw] = (_Float16)p[1]; A_v[2048 + sw] = (_Float16)p[2];
  }
  for (int idx = lane; idx < 256; idx += 32) {
    int el = idx >> 4, m = idx & 15;
    const float* p = hn_v + edst[el] * 48 + m * 3;
    int sw = a_swz(el, 32 + m);
    A_v[sw] = (_Float16)p[0]; A_v[1024 + sw] = (_Float16)p[1]; A_v[2048 + sw] = (_Float16)p[2];
  }
  for (int idx = lane; idx < 64; idx += 32) {  // zero-pad k = 48..63
    int el = idx >> 2, c = 48 + (idx & 3) * 4;
    store4h(A_v, el, c, 0.f, 0.f, 0.f, 0.f);
    store4h(A_v + 1024, el, c, 0.f, 0.f, 0.f, 0.f);
    store4h(A_v + 2048, el, c, 0.f, 0.f, 0.f, 0.f);
  }
  __syncthreads();

  // ---- stage 1 glin (ev): hs tiles 0..7 -> SiLU, contract with Ls -> s1 ----
  {
    float accs[8];
#pragma unroll
    for (int r = 0; r < 8; ++r) accs[r] = 0.f;
    for (int t = 0; t < 8; ++t) {
      v8f acc = mm_tile<3>(A_s, Wevs, t);
      float L = Levs[t * 16 + n16];
#pragma unroll
      for (int r = 0; r < 8; ++r) {
        float x = acc[r];
        accs[r] += x * sgm(x) * L;
      }
    }
#pragma unroll
    for (int r = 0; r < 8; ++r) {
      float tot = reduce16(accs[r]);
      if (n16 == 0) s1[r + 8 * half] = tot;
    }
  }
  // hs tiles 8..11 -> sigmoid gates
  for (int t = 8; t < 12; ++t) {
    v8f acc = mm_tile<3>(A_s, Wevs, t);
#pragma unroll
    for (int r = 0; r < 8; ++r)
      gate[(r + 8 * half) * 64 + (t - 8) * 16 + n16] = sgm(acc[r]);
  }
  __syncthreads();

  // hv per component, gated contraction with Lv -> v1
#pragma unroll
  for (int comp = 0; comp < 3; ++comp) {
    float accv[8];
#pragma unroll
    for (int r = 0; r < 8; ++r) accv[r] = 0.f;
    for (int t = 0; t < 4; ++t) {
      v8f acc = mm_tile<2>(A_v + comp * 1024, Wevv, t);
      int k = t * 16 + n16;
      float L = Levv[k];
#pragma unroll
      for (int r = 0; r < 8; ++r)
        accv[r] += acc[r] * gate[(r + 8 * half) * 64 + k] * L;
    }
#pragma unroll
    for (int r = 0; r < 8; ++r) {
      float tot = reduce16(accv[r]);
      if (n16 == 0) v1[(r + 8 * half) * 3 + comp] = tot;
    }
  }
  __syncthreads();

  // ---- fc MLP: w = relu(fes@W1*0.25)@W2*0.25 via WMMA (K padded) ----
  for (int idx = lane; idx < 128; idx += 32) {
    int el = idx >> 3, ch = idx & 7;
    int e = e0 + el; if (e >= NE) e = NE - 1;
    if (ch < 4) {
      float4 v = ((const float4*)(fes + e * 16))[ch];
      store4h(fesA, el, ch * 4, v.x, v.y, v.z, v.w);
    } else {
      store4h(fesA, el, ch * 4, 0.f, 0.f, 0.f, 0.f);
    }
  }
  __syncthreads();
  {
    v8f acc = mm_tile<1>(fesA, Wfc1, 0);
#pragma unroll
    for (int r = 0; r < 8; ++r) {
      int el = r + 8 * half;
      relA[a_swz(el, n16)] = (_Float16)fmaxf(acc[r], 0.f);
      relA[a_swz(el, 16 + n16)] = (_Float16)0.f;
    }
  }
  __syncthreads();
  for (int t = 0; t < 6; ++t) {
    v8f acc = mm_tile<1>(relA, Wfc2, t);
#pragma unroll
    for (int r = 0; r < 8; ++r)
      wbuf[(r + 8 * half) * 96 + t * 16 + n16] = acc[r];
  }
  __syncthreads();

  // ---- tensor product (VALU); overwrite A staging in place ----
  const float c3 = 0.57735026918962576f, c2 = 0.70710678118654752f;
  {
    int c = lane; // 32 scalar channels
    for (int el = 0; el < 16; ++el) {
      float S1 = s1[el], S2 = s2v[el];
      float dv = v1[el * 3 + 0] * v2v[el * 3 + 0] + v1[el * 3 + 1] * v2v[el * 3 + 1] +
                 v1[el * 3 + 2] * v2v[el * 3 + 2];
      float ts = (wbuf[el * 96 + c] * (S1 * S2) + wbuf[el * 96 + 32 + c] * (dv * c3)) * c2;
      A_s[a_swz(el, c)] = (_Float16)ts;
    }
  }
  if (lane < 16) {
    int m = lane; // 16 vector channels
    for (int el = 0; el < 16; ++el) {
      float S1 = s1[el], S2 = s2v[el];
      float wsv = wbuf[el * 96 + 64 + m], wvs = wbuf[el * 96 + 80 + m];
#pragma unroll
      for (int comp = 0; comp < 3; ++comp) {
        float tv = (wsv * S1 * v2v[el * 3 + comp] + wvs * S2 * v1[el * 3 + comp]) * (c3 * c2);
        A_v[comp * 1024 + a_swz(el, m)] = (_Float16)tv;
      }
    }
  }
  __syncthreads();

  // ---- stage 2 glin (eu) ----
  for (int t = 0; t < 8; ++t) {
    v8f acc = mm_tile<3>(A_s, Weus, t);
#pragma unroll
    for (int r = 0; r < 8; ++r) {
      float x = acc[r];
      g2s[a_swz(r + 8 * half, t * 16 + n16)] = (_Float16)(x * sgm(x));
    }
  }
  for (int t = 8; t < 12; ++t) {
    v8f acc = mm_tile<3>(A_s, Weus, t);
#pragma unroll
    for (int r = 0; r < 8; ++r)
      gate[(r + 8 * half) * 64 + (t - 8) * 16 + n16] = sgm(acc[r]);
  }
  __syncthreads();
#pragma unroll
  for (int comp = 0; comp < 3; ++comp) {
    for (int t = 0; t < 4; ++t) {
      v8f acc = mm_tile<2>(A_v + comp * 1024, Weuv, t);
#pragma unroll
      for (int r = 0; r < 8; ++r) {
        int el = r + 8 * half;
        int k = t * 16 + n16;
        g2v[comp * 1024 + a_swz(el, k)] = (_Float16)(acc[r] * gate[el * 64 + k]);
      }
    }
  }
  __syncthreads();

  // ---- final projections, outputs, segment-sum atomics ----
  for (int t = 0; t < 2; ++t) {
    v8f acc = mm_tile<4>(g2s, Wels, t);
#pragma unroll
    for (int r = 0; r < 8; ++r) {
      int el = r + 8 * half;
      int e = e0 + el;
      if (e < NE) {
        int c = t * 16 + n16;
        float v = he_s[e * 32 + c] + acc[r];
        o_he_s[e * 32 + c] = v;
        atomicAdd(&nf_s[edst[el] * 32 + c], v * nrm[el]);
      }
    }
  }
#pragma unroll
  for (int comp = 0; comp < 3; ++comp) {
    v8f acc = mm_tile<2>(g2v + comp * 1024, Welv, 0);
#pragma unroll
    for (int r = 0; r < 8; ++r) {
      int el = r + 8 * half;
      int e = e0 + el;
      if (e < NE) {
        int m = n16;
        float v = he_v[(e * 16 + m) * 3 + comp] + acc[r];
        o_he_v[(e * 16 + m) * 3 + comp] = v;
        atomicAdd(&nf_v[(edst[el] * 16 + m) * 3 + comp], v * nrm[el]);
      }
    }
  }
}

// ======================= NODE UPDATE KERNEL =======================
__global__ void __launch_bounds__(256) node_kernel(
    const float* __restrict__ hn_s, const float* __restrict__ hn_v,
    const float* __restrict__ nl_Ws, const float* __restrict__ nl_Wv,
    const float* __restrict__ nl_Ls, const float* __restrict__ nl_Lv,
    const float* __restrict__ nf_s, const float* __restrict__ nf_v,
    float* __restrict__ o_s, float* __restrict__ o_v) {
  extern __shared__ char smem[];
  _Float16* Ws = (_Float16*)smem;            // 64x192: KT=2 NT=12 (24576 B)
  _Float16* Wv = (_Float16*)(smem + 24576);  // 32x64:  KT=1 NT=4  (4096 B)
  _Float16* Ls = (_Float16*)(smem + 28672);  // 128x32: KT=4 NT=2  (8192 B)
  _Float16* Lv = (_Float16*)(smem + 36864);  // 64x16:  KT=2 NT=1  (2048 B)
  char* wavebase = smem + 38912;

  const int wid = threadIdx.x >> 5;
  const int lane = threadIdx.x & 31;
  const int n16 = lane & 15;
  const int half = lane >> 4;
  char* wa = wavebase + wid * 19456;
  _Float16* A_s  = (_Float16*)wa;            // 2 tiles (16x64)
  _Float16* A_v  = (_Float16*)(wa + 2048);   // 3 comps x 1 tile (16x32)
  float*    gate = (float*)(wa + 5120);      // 16x64
  _Float16* g2s  = (_Float16*)(wa + 9216);   // 4 tiles (16x128)
  _Float16* g2v  = (_Float16*)(wa + 13312);  // 3 comps x 2 tiles (16x64)

  const float sW = rsqrtf(64.f), sV = rsqrtf(32.f), sLs = rsqrtf(128.f), sLv = rsqrtf(64.f);
  fill_b(Ws, nl_Ws, 64, 192, 2, 12, sW);
  fill_b(Wv, nl_Wv, 32, 64, 1, 4, sV);
  fill_b(Ls, nl_Ls, 128, 32, 4, 2, sLs);
  fill_b(Lv, nl_Lv, 64, 16, 2, 1, sLv);
  __syncthreads();

  const int n0 = blockIdx.x * 128 + wid * 16;
  for (int idx = lane; idx < 128; idx += 32) {
    int nl = idx >> 3, ch = idx & 7;
    int n = n0 + nl; if (n >= NN) n = NN - 1;
    float4 v = ((const float4*)(hn_s + n * 32))[ch];
    store4h(A_s, nl, ch * 4, v.x, v.y, v.z, v.w);
  }
  for (int idx = lane; idx < 128; idx += 32) {
    int nl = idx >> 3, ch = idx & 7;
    int n = n0 + nl; if (n >= NN) n = NN - 1;
    float4 v = ((const float4*)(nf_s + n * 32))[ch];
    store4h(A_s, nl, 32 + ch * 4, v.x, v.y, v.z, v.w);
  }
  for (int idx = lane; idx < 256; idx += 32) {
    int nl = idx >> 4, m = idx & 15;
    int n = n0 + nl; if (n >= NN) n = NN - 1;
    const float* p = hn_v + n * 48 + m * 3;
    int sw = a_swz(nl, m);
    A_v[sw] = (_Float16)p[0]; A_v[512 + sw] = (_Float16)p[1]; A_v[1024 + sw] = (_Float16)p[2];
  }
  for (int idx = lane; idx < 256; idx += 32) {
    int nl = idx >> 4, m = idx & 15;
    int n = n0 + nl; if (n >= NN) n = NN - 1;
    const float* p = nf_v + n * 48 + m * 3;
    int sw = a_swz(nl, 16 + m);
    A_v[sw] = (_Float16)p[0]; A_v[512 + sw] = (_Float16)p[1]; A_v[1024 + sw] = (_Float16)p[2];
  }
  __syncthreads();

  for (int t = 0; t < 8; ++t) {
    v8f acc = mm_tile<2>(A_s, Ws, t);
#pragma unroll
    for (int r = 0; r < 8; ++r) {
      float x = acc[r];
      g2s[a_swz(r + 8 * half, t * 16 + n16)] = (_Float16)(x * sgm(x));
    }
  }
  for (int t = 8; t < 12; ++t) {
    v8f acc = mm_tile<2>(A_s, Ws, t);
#pragma unroll
    for (int r = 0; r < 8; ++r)
      gate[(r + 8 * half) * 64 + (t - 8) * 16 + n16] = sgm(acc[r]);
  }
  __syncthreads();
#pragma unroll
  for (int comp = 0; comp < 3; ++comp) {
    for (int t = 0; t < 4; ++t) {
      v8f acc = mm_tile<1>(A_v + comp * 512, Wv, t);
#pragma unroll
      for (int r = 0; r < 8; ++r) {
        int nl = r + 8 * half;
        int k = t * 16 + n16;
        g2v[comp * 1024 + a_swz(nl, k)] = (_Float16)(acc[r] * gate[nl * 64 + k]);
      }
    }
  }
  __syncthreads();

  for (int t = 0; t < 2; ++t) {
    v8f acc = mm_tile<4>(g2s, Ls, t);
#pragma unroll
    for (int r = 0; r < 8; ++r) {
      int n = n0 + r + 8 * half;
      if (n < NN) {
        int c = t * 16 + n16;
        o_s[n * 32 + c] = hn_s[n * 32 + c] + acc[r];
      }
    }
  }
#pragma unroll
  for (int comp = 0; comp < 3; ++comp) {
    v8f acc = mm_tile<2>(g2v + comp * 1024, Lv, 0);
#pragma unroll
    for (int r = 0; r < 8; ++r) {
      int n = n0 + r + 8 * half;
      if (n < NN) {
        int m = n16;
        o_v[(n * 16 + m) * 3 + comp] = hn_v[(n * 16 + m) * 3 + comp] + acc[r];
      }
    }
  }
}

extern "C" void kernel_launch(void* const* d_in, const int* in_sizes, int n_in,
                              void* d_out, int out_size, void* d_ws, size_t ws_size,
                              hipStream_t stream) {
  const int*   eidx  = (const int*)d_in[0];
  const float* hn_s  = (const float*)d_in[1];
  const float* hn_v  = (const float*)d_in[2];
  const float* he_s  = (const float*)d_in[3];
  const float* he_v  = (const float*)d_in[4];
  const float* fe_s  = (const float*)d_in[5];
  const float* fe_v  = (const float*)d_in[6];
  const float* fes   = (const float*)d_in[7];
  const float* norm  = (const float*)d_in[8];
  const float* ev_Ws = (const float*)d_in[9];
  const float* ev_Wv = (const float*)d_in[10];
  const float* ev_Ls = (const float*)d_in[11];
  const float* ev_Lv = (const float*)d_in[12];
  const float* fc_W1 = (const float*)d_in[13];
  const float* fc_W2 = (const float*)d_in[14];
  const float* eu_Ws = (const float*)d_in[15];
  const float* eu_Wv = (const float*)d_in[16];
  const float* eu_Ls = (const float*)d_in[17];
  const float* eu_Lv = (const float*)d_in[18];
  const float* nl_Ws = (const float*)d_in[19];
  const float* nl_Wv = (const float*)d_in[20];
  const float* nl_Ls = (const float*)d_in[21];
  const float* nl_Lv = (const float*)d_in[22];

  float* out = (float*)d_out;
  float* o_hn_s = out;                                         // N x 32
  float* o_hn_v = out + (size_t)NN * 32;                       // N x 16 x 3
  float* o_he_s = out + (size_t)NN * 80;                       // E x 32
  float* o_he_v = out + (size_t)NN * 80 + (size_t)NE * 32;     // E x 16 x 3

  float* nf_s = (float*)d_ws;                                  // N x 32
  float* nf_v = nf_s + (size_t)NN * 32;                        // N x 16 x 3

  hipMemsetAsync(d_ws, 0, (size_t)NN * 80 * sizeof(float), stream);

  const size_t smem_edge = 108288 + 4 * 24320;  // ~201 KB
  const size_t smem_node = 38912 + 8 * 19456;   // ~190 KB
  hipFuncSetAttribute((const void*)edge_kernel,
                      hipFuncAttributeMaxDynamicSharedMemorySize, (int)smem_edge);
  hipFuncSetAttribute((const void*)node_kernel,
                      hipFuncAttributeMaxDynamicSharedMemorySize, (int)smem_node);

  edge_kernel<<<(NE + 63) / 64, 128, smem_edge, stream>>>(
      eidx, hn_s, hn_v, he_s, he_v, fe_s, fe_v, fes, norm,
      ev_Ws, ev_Wv, ev_Ls, ev_Lv, fc_W1, fc_W2,
      eu_Ws, eu_Wv, eu_Ls, eu_Lv,
      o_he_s, o_he_v, nf_s, nf_v);

  node_kernel<<<(NN + 127) / 128, 256, smem_node, stream>>>(
      hn_s, hn_v, nl_Ws, nl_Wv, nl_Ls, nl_Lv, nf_s, nf_v, o_hn_s, o_hn_v);
}
